// TransformerConv_17549236371610
// MI455X (gfx1250) — matrix operand order repaired and checked by reading, not compiled, not run on hardware
//
#include <hip/hip_runtime.h>
#include <hip/hip_bf16.h>

// ---------------------------------------------------------------------------
// Problem constants (match reference)
// ---------------------------------------------------------------------------
constexpr int BG  = 8;        // graphs
constexpr int NN  = 1024;     // nodes per graph
constexpr int CC  = 128;      // channels
constexpr int HH  = 8;        // heads
constexpr int DH  = 16;       // head dim
constexpr int EE  = 262144;   // edges
constexpr int TOT = BG * NN;  // 8192 nodes
constexpr float EPSBN = 1e-5f;

typedef __attribute__((ext_vector_type(16))) __bf16 v16bf;
typedef __attribute__((ext_vector_type(8)))  float  v8f;
typedef __attribute__((address_space(3))) unsigned char lds_uchar;

// ---------------------------------------------------------------------------
// WMMA fragment loaders (wave32; A and B use the same contiguous k order)
//   A (16x32 bf16): lane<16 -> row=lane, K elems 0..15 (contiguous)
//                   lane>=16 -> row=lane-16, K elems 16..31
//   B (32x16 bf16): lane<16 -> col=lane, K elems 0..15 (contiguous)
//                   lane>=16 -> col=lane-16, K elems 16..31
//   C/D (16x16 f32): lane -> col (lane&15); vgpr r -> row r (+8 for lane>=16)
// ---------------------------------------------------------------------------
__device__ __forceinline__ v16bf loadA_rm(const __bf16* A, int ld, int row0, int k0) {
  const int lane = threadIdx.x & 31;
  const int m = lane & 15, hi = lane >> 4;
  const __bf16* p = A + (size_t)(row0 + m) * ld + k0 + 16 * hi;
  v16bf f;
#pragma unroll
  for (int i = 0; i < 8; ++i) f[i] = p[i];          // K = k0+16hi .. +7
#pragma unroll
  for (int i = 0; i < 8; ++i) f[8 + i] = p[8 + i];  // K = k0+16hi+8 .. +7
  return f;
}

// B = W^T where W is row-major [Ncols x K]; column n of B is row n of W.
__device__ __forceinline__ v16bf loadB_Wt(const __bf16* W, int ld, int n0, int k0) {
  const int lane = threadIdx.x & 31;
  const int n = lane & 15, hi = lane >> 4;
  const __bf16* p = W + (size_t)(n0 + n) * ld + k0 + 16 * hi;
  v16bf f;
#pragma unroll
  for (int i = 0; i < 16; ++i) f[i] = p[i];
  return f;
}

__device__ __forceinline__ v8f wmma_bf16(v16bf a, v16bf b, v8f c) {
  return __builtin_amdgcn_wmma_f32_16x16x32_bf16(false, a, false, b, (short)0, c,
                                                 false, false);
}

// ---------------------------------------------------------------------------
// Small helper kernels
// ---------------------------------------------------------------------------
__global__ void cvt_f32_bf16(const float* __restrict__ in, __bf16* __restrict__ out, int n) {
  int i = blockIdx.x * blockDim.x + threadIdx.x;
  if (i < n) out[i] = (__bf16)in[i];
}

__global__ void adj_scatter(const int* __restrict__ ei, const int* __restrict__ et,
                            unsigned char* __restrict__ adj) {
  int i = blockIdx.x * blockDim.x + threadIdx.x;
  if (i >= EE) return;
  int s = ei[i], d = ei[EE + i];
  int g = s >> 10, ls = s & 1023, ld = d & 1023;
  adj[((size_t)g * NN + ls) * NN + ld] = (unsigned char)(et[i] + 1);
}

__global__ void adj_diag(unsigned char* __restrict__ adj) {
  int i = blockIdx.x * blockDim.x + threadIdx.x; // 0..TOT-1
  if (i >= TOT) return;
  int g = i >> 10, nn = i & 1023;
  unsigned char* p = adj + ((size_t)g * NN + nn) * NN + nn;
  if (*p == 0) *p = 22;
}

// ---------------------------------------------------------------------------
// QKV projection: xb[8192x128] @ in_proj_w^T [128x384] + b.
// Each wave computes a 64-row x 16-col strip; k-loop double-buffered so next
// step's fragments load under the current WMMA burst.
// Q,K stored head-major [B*H][N][DH]; V stored TRANSPOSED [B*H][DH][N] so the
// attention B-fragment load is contiguous.
// grid (TOT/64, 384/64), block 128 (4 waves).
// ---------------------------------------------------------------------------
__global__ __launch_bounds__(128) void gemm_qkv(
    const __bf16* __restrict__ xb, const __bf16* __restrict__ wb,
    const float* __restrict__ bias,
    __bf16* __restrict__ Q, __bf16* __restrict__ K, __bf16* __restrict__ Vt) {
  const int wave = threadIdx.x >> 5;
  const int row0 = blockIdx.x * 64;
  const int col0 = (blockIdx.y * 4 + wave) * 16;
  v8f acc[4] = {{}, {}, {}, {}};
  v16bf bcur = loadB_Wt(wb, CC, col0, 0);
  v16bf acur[4];
#pragma unroll
  for (int t = 0; t < 4; ++t) acur[t] = loadA_rm(xb, CC, row0 + 16 * t, 0);
#pragma unroll
  for (int k0 = 0; k0 < CC; k0 += 32) {
    v16bf bnx = bcur;
    v16bf anx[4] = {acur[0], acur[1], acur[2], acur[3]};
    if (k0 + 32 < CC) {
      bnx = loadB_Wt(wb, CC, col0, k0 + 32);
#pragma unroll
      for (int t = 0; t < 4; ++t) anx[t] = loadA_rm(xb, CC, row0 + 16 * t, k0 + 32);
    }
#pragma unroll
    for (int t = 0; t < 4; ++t) acc[t] = wmma_bf16(acur[t], bcur, acc[t]);
    bcur = bnx;
#pragma unroll
    for (int t = 0; t < 4; ++t) acur[t] = anx[t];
  }
  const int lane = threadIdx.x & 31;
  const int n = lane & 15, hi = lane >> 4;
  const int c = col0 + n;
  const int which = c >> 7;     // 0=q 1=k 2=v
  const int cc = c & 127;
  const int head = cc >> 4, d = cc & 15;
  const float bv = bias[c];
#pragma unroll
  for (int t = 0; t < 4; ++t) {
#pragma unroll
    for (int r = 0; r < 8; ++r) {
      int node = row0 + 16 * t + r + 8 * hi;  // 0..8191
      int b_ = node >> 10, nn = node & 1023;
      int bh = b_ * HH + head;
      __bf16 v = (__bf16)(acc[t][r] + bv);
      if (which == 0)      Q [((size_t)bh * NN + nn) * DH + d] = v;
      else if (which == 1) K [((size_t)bh * NN + nn) * DH + d] = v;
      else                 Vt[((size_t)bh * DH + d) * NN + nn] = v;
    }
  }
}

// ---------------------------------------------------------------------------
// Flash attention: one wave handles 16 query rows of one (b,h).
// k-step 32: two QK^T WMMAs (K-dim padded 16->32); 16x32 adjacency tile
// fetched global->LDS with CDNA5 async-to-LDS (one b128 per lane, completed
// via s_wait_asynccnt); bias table gathered from LDS; online softmax; P
// transposed via per-wave LDS tile; one PV WMMA (V transposed layout).
// grid = B*H*(N/16)/4 blocks, block 128.
// ---------------------------------------------------------------------------
__global__ __launch_bounds__(128) void attn_flash(
    const __bf16* __restrict__ Q, const __bf16* __restrict__ K,
    const __bf16* __restrict__ Vt, const unsigned char* __restrict__ adj,
    const float* __restrict__ table, __bf16* __restrict__ O) {
  const int wave = threadIdx.x >> 5;
  const int lane = threadIdx.x & 31;
  const int gtile = blockIdx.x * 4 + wave;
  const int qt = gtile & 63;             // 64 q-tiles per (b,h)
  const int bh = gtile >> 6;             // 0..63
  const int b = bh >> 3, h = bh & 7;
  const __bf16* Qp  = Q  + (size_t)bh * NN * DH;
  const __bf16* Kp  = K  + (size_t)bh * NN * DH;
  const __bf16* Vtp = Vt + (size_t)bh * DH * NN;
  const unsigned char* adjp = adj + (size_t)b * NN * NN;
  const int q0 = qt * 16;
  const int n = lane & 15, hi = lane >> 4;

  __shared__ float tbl[23 * HH];                       // bias table
  __shared__ __align__(16) unsigned char adjt[4][16 * 32];
  __shared__ __bf16 plds[4][16 * 32];
  for (int i = threadIdx.x; i < 23 * HH; i += 128) tbl[i] = table[i];
  __syncthreads();
  unsigned char* myadj = adjt[wave];
  __bf16* myp = plds[wave];

  // Q fragment: valid K-dim = DH = 16; upper half zero.
  v16bf qf;
  {
    const __bf16* p = Qp + (size_t)(q0 + n) * DH;
#pragma unroll
    for (int i = 0; i < 16; ++i) qf[i] = hi ? (__bf16)0.0f : p[i];
  }

  float mrow[8], lrow[8];
#pragma unroll
  for (int r = 0; r < 8; ++r) { mrow[r] = -1e30f; lrow[r] = 0.0f; }
  v8f oacc = {};

  for (int k0 = 0; k0 < NN; k0 += 32) {
    // ---- stage 16x32 u8 adjacency tile: async global->LDS, one b128/lane ----
    {
      lds_uchar* dstl = (lds_uchar*)(myadj + n * 32 + 16 * hi);
      const unsigned char* gsrc = adjp + (size_t)(q0 + n) * NN + k0 + 16 * hi;
      asm volatile("global_load_async_to_lds_b128 %0, %1, off"
                   :: "v"(dstl), "v"(gsrc)
                   : "memory");
      asm volatile("s_wait_asynccnt 0" ::: "memory");
    }

    // ---- scores: two 16x16 tiles ----
    v8f s0 = {}, s1 = {};
#pragma unroll
    for (int sub = 0; sub < 2; ++sub) {
      v16bf kf;
      if (hi == 0) {  // B col n = K row (k0+16*sub+n), d = 0..15 contiguous
        const __bf16* p = Kp + (size_t)(k0 + sub * 16 + n) * DH;
#pragma unroll
        for (int i = 0; i < 16; ++i) kf[i] = p[i];
      } else {        // K elems 16..31 of B are the zero-padded d>=16
#pragma unroll
        for (int i = 0; i < 16; ++i) kf[i] = (__bf16)0.0f;
      }
      if (sub == 0) s0 = wmma_bf16(qf, kf, s0);
      else          s1 = wmma_bf16(qf, kf, s1);
    }

    // ---- scale + bias lookup (adjacency tile + table both in LDS) ----
    float sv0[8], sv1[8];
#pragma unroll
    for (int r = 0; r < 8; ++r) {
      int rr = (r + 8 * hi) * 32;
      sv0[r] = s0[r] * 0.25f + tbl[(int)myadj[rr + n] * HH + h];
      sv1[r] = s1[r] * 0.25f + tbl[(int)myadj[rr + 16 + n] * HH + h];
    }

    // ---- online softmax update ----
#pragma unroll
    for (int r = 0; r < 8; ++r) {
      float v = fmaxf(sv0[r], sv1[r]);
#pragma unroll
      for (int off = 1; off < 16; off <<= 1) v = fmaxf(v, __shfl_xor(v, off, 16));
      float mnew = fmaxf(mrow[r], v);
      float scale = __expf(mrow[r] - mnew);
      float p0 = __expf(sv0[r] - mnew);
      float p1 = __expf(sv1[r] - mnew);
      float rs = p0 + p1;
#pragma unroll
      for (int off = 1; off < 16; off <<= 1) rs += __shfl_xor(rs, off, 16);
      lrow[r] = lrow[r] * scale + rs;
      mrow[r] = mnew;
      oacc[r] *= scale;
      sv0[r] = p0; sv1[r] = p1;
    }

    // ---- transpose P (C layout -> A fragment) via private LDS tile ----
#pragma unroll
    for (int r = 0; r < 8; ++r) {
      myp[(r + 8 * hi) * 32 + n]      = (__bf16)sv0[r];
      myp[(r + 8 * hi) * 32 + 16 + n] = (__bf16)sv1[r];
    }
    // same-wave DS ops are in-order; compiler inserts s_wait_dscnt for RAW
    v16bf pf;
    {
      const __bf16* p = myp + n * 32 + 16 * hi;
#pragma unroll
      for (int i = 0; i < 8; ++i) pf[i] = p[i];
#pragma unroll
      for (int i = 0; i < 8; ++i) pf[8 + i] = p[8 + i];
    }

    // ---- V fragment: B[k][d] = Vt[d][k0+k] -> contiguous per lane ----
    v16bf vf;
    {
      const __bf16* p = Vtp + (size_t)n * NN + k0 + 16 * hi;
#pragma unroll
      for (int j = 0; j < 16; ++j) vf[j] = p[j];
    }
    oacc = wmma_bf16(pf, vf, oacc);
  }

  // ---- finalize: O[node][h*16+d] ----
#pragma unroll
  for (int r = 0; r < 8; ++r) {
    float v = oacc[r] / lrow[r];
    int node = b * NN + q0 + r + 8 * hi;
    O[(size_t)node * CC + h * DH + n] = (__bf16)v;
  }
}

// ---------------------------------------------------------------------------
// Generic bf16 GEMM:  out[M x Nc] = A[M x Kd] @ W[Nc x Kd]^T + bias
// Each wave computes a 64-row x 16-col strip; k-loop double-buffered.
// optional relu / f32 residual / f32 + bf16 stores.
// grid (M/64, Nc/64), block 128.
// ---------------------------------------------------------------------------
template <bool RELU>
__global__ __launch_bounds__(128) void gemm_generic(
    const __bf16* __restrict__ A, const __bf16* __restrict__ W,
    const float* __restrict__ bias, const float* __restrict__ res,
    float* __restrict__ outF, __bf16* __restrict__ outB, int Kd, int Nc) {
  const int wave = threadIdx.x >> 5;
  const int row0 = blockIdx.x * 64;
  const int col0 = (blockIdx.y * 4 + wave) * 16;
  v8f acc[4] = {{}, {}, {}, {}};
  v16bf bcur = loadB_Wt(W, Kd, col0, 0);
  v16bf acur[4];
#pragma unroll
  for (int t = 0; t < 4; ++t) acur[t] = loadA_rm(A, Kd, row0 + 16 * t, 0);
  for (int k0 = 0; k0 < Kd; k0 += 32) {
    v16bf bnx = bcur;
    v16bf anx[4] = {acur[0], acur[1], acur[2], acur[3]};
    if (k0 + 32 < Kd) {
      bnx = loadB_Wt(W, Kd, col0, k0 + 32);
#pragma unroll
      for (int t = 0; t < 4; ++t) anx[t] = loadA_rm(A, Kd, row0 + 16 * t, k0 + 32);
    }
#pragma unroll
    for (int t = 0; t < 4; ++t) acc[t] = wmma_bf16(acur[t], bcur, acc[t]);
    bcur = bnx;
#pragma unroll
    for (int t = 0; t < 4; ++t) acur[t] = anx[t];
  }
  const int lane = threadIdx.x & 31;
  const int n = lane & 15, hi = lane >> 4;
  const float bv = bias[col0 + n];
#pragma unroll
  for (int t = 0; t < 4; ++t) {
#pragma unroll
    for (int r = 0; r < 8; ++r) {
      int row = row0 + 16 * t + r + 8 * hi;
      size_t idx = (size_t)row * Nc + col0 + n;
      float v = acc[t][r] + bv;
      if (RELU) v = fmaxf(v, 0.0f);
      if (res)  v += res[idx];
      if (outF) outF[idx] = v;
      if (outB) outB[idx] = (__bf16)v;
    }
  }
}

// ---------------------------------------------------------------------------
// BatchNorm1d (training-mode stats over all rows) -- coalesced partials.
// bn_partial: grid 64 blocks x 256 threads; block covers 128 rows; each wave
// reads contiguous 128B lines. Per-channel partials atomicAdd'ed to global.
// ---------------------------------------------------------------------------
__global__ __launch_bounds__(256) void bn_partial(const float* __restrict__ t,
                                                  float* __restrict__ sums,
                                                  float* __restrict__ sumsq) {
  const int ch = threadIdx.x & 127;
  const int half = threadIdx.x >> 7;   // 0 or 1
  const int rbase = blockIdx.x * 128;
  float s = 0.0f, ss = 0.0f;
  for (int r = half; r < 128; r += 2) {
    float v = t[(size_t)(rbase + r) * CC + ch];
    s += v; ss += v * v;
  }
  __shared__ float sh0[256], sh1[256];
  sh0[threadIdx.x] = s; sh1[threadIdx.x] = ss;
  __syncthreads();
  if (half == 0) {
    s += sh0[threadIdx.x + 128];
    ss += sh1[threadIdx.x + 128];
    atomicAdd(&sums[ch], s);
    atomicAdd(&sumsq[ch], ss);
  }
}

__global__ void bn_finalize(const float* __restrict__ sums, const float* __restrict__ sumsq,
                            float* __restrict__ mu, float* __restrict__ rsig) {
  int c = threadIdx.x;  // 128 threads, 1 block
  float m = sums[c] / (float)TOT;
  float var = sumsq[c] / (float)TOT - m * m;
  mu[c] = m;
  rsig[c] = rsqrtf(var + EPSBN);
}

__global__ void bn_apply(const float* __restrict__ t, const float* __restrict__ mu,
                         const float* __restrict__ rsig, const float* __restrict__ gamma,
                         const float* __restrict__ beta, float* __restrict__ outF,
                         __bf16* __restrict__ outB) {
  int i = blockIdx.x * blockDim.x + threadIdx.x;
  if (i >= TOT * CC) return;
  int c = i & (CC - 1);
  float v = (t[i] - mu[c]) * rsig[c] * gamma[c] + beta[c];
  if (outF) outF[i] = v;
  if (outB) outB[i] = (__bf16)v;
}

// ---------------------------------------------------------------------------
// Host orchestration
// ---------------------------------------------------------------------------
static inline size_t alignUp(size_t x) { return (x + 255) & ~(size_t)255; }

extern "C" void kernel_launch(void* const* d_in, const int* in_sizes, int n_in,
                              void* d_out, int out_size, void* d_ws, size_t ws_size,
                              hipStream_t stream) {
  const float* x            = (const float*)d_in[0];
  const int*   edge_index   = (const int*)d_in[1];
  const int*   edge_type    = (const int*)d_in[2];
  // d_in[3] = batch (unused; derived from node index)
  const float* bias_table   = (const float*)d_in[4];
  const float* in_proj_w    = (const float*)d_in[5];
  const float* in_proj_b    = (const float*)d_in[6];
  const float* out_proj_w   = (const float*)d_in[7];
  const float* out_proj_b   = (const float*)d_in[8];
  const float* mlp_w1       = (const float*)d_in[9];
  const float* mlp_b1       = (const float*)d_in[10];
  const float* mlp_w2       = (const float*)d_in[11];
  const float* mlp_b2       = (const float*)d_in[12];
  const float* n2g          = (const float*)d_in[13];
  const float* n2b          = (const float*)d_in[14];
  const float* n3g          = (const float*)d_in[15];
  const float* n3b          = (const float*)d_in[16];
  float* out = (float*)d_out;

  char* ws = (char*)d_ws;
  size_t o = 0;
  auto take = [&](size_t bytes) { char* p = ws + o; o += alignUp(bytes); return p; };

  __bf16* xb   = (__bf16*)take((size_t)TOT * CC * 2);
  __bf16* wqkv = (__bf16*)take((size_t)3 * CC * CC * 2);
  __bf16* wob  = (__bf16*)take((size_t)CC * CC * 2);
  __bf16* w1b  = (__bf16*)take((size_t)2 * CC * CC * 2);
  __bf16* w2b  = (__bf16*)take((size_t)CC * 2 * CC * 2);
  __bf16* Qb   = (__bf16*)take((size_t)BG * HH * NN * DH * 2);
  __bf16* Kb   = (__bf16*)take((size_t)BG * HH * NN * DH * 2);
  __bf16* Vtb  = (__bf16*)take((size_t)BG * HH * DH * NN * 2);
  unsigned char* adj = (unsigned char*)take((size_t)BG * NN * NN);
  __bf16* Ob   = (__bf16*)take((size_t)TOT * CC * 2);
  float*  t1   = (float*)take((size_t)TOT * CC * 4);
  float*  h2   = (float*)take((size_t)TOT * CC * 4);
  __bf16* h2b  = (__bf16*)take((size_t)TOT * CC * 2);
  __bf16* m1b  = (__bf16*)take((size_t)TOT * 2 * CC * 2);
  float*  t2   = (float*)take((size_t)TOT * CC * 4);
  float*  sums2  = (float*)take(CC * 4);
  float*  sumsq2 = (float*)take(CC * 4);
  float*  sums3  = (float*)take(CC * 4);
  float*  sumsq3 = (float*)take(CC * 4);
  float*  mu2  = (float*)take(CC * 4);
  float*  rs2  = (float*)take(CC * 4);
  float*  mu3  = (float*)take(CC * 4);
  float*  rs3  = (float*)take(CC * 4);
  (void)ws_size; (void)n_in; (void)in_sizes; (void)out_size;

  // 0. zero accumulators (fresh every call -> deterministic under replay)
  hipMemsetAsync(sums2, 0, 4 * CC * 4, stream);  // sums2..sumsq3 contiguous

  // 1. dense adjacency (edge_type + 1, self-loop fill 22)
  hipMemsetAsync(adj, 0, (size_t)BG * NN * NN, stream);
  adj_scatter<<<(EE + 255) / 256, 256, 0, stream>>>(edge_index, edge_type, adj);
  adj_diag<<<(TOT + 255) / 256, 256, 0, stream>>>(adj);

  // 2. bf16 conversions
  cvt_f32_bf16<<<(TOT * CC + 255) / 256, 256, 0, stream>>>(x, xb, TOT * CC);
  cvt_f32_bf16<<<(3 * CC * CC + 255) / 256, 256, 0, stream>>>(in_proj_w, wqkv, 3 * CC * CC);
  cvt_f32_bf16<<<(CC * CC + 255) / 256, 256, 0, stream>>>(out_proj_w, wob, CC * CC);
  cvt_f32_bf16<<<(2 * CC * CC + 255) / 256, 256, 0, stream>>>(mlp_w1, w1b, 2 * CC * CC);
  cvt_f32_bf16<<<(2 * CC * CC + 255) / 256, 256, 0, stream>>>(mlp_w2, w2b, 2 * CC * CC);

  // 3. QKV projection (WMMA) -> head-major Q/K, transposed V
  gemm_qkv<<<dim3(TOT / 64, (3 * CC) / 64), 128, 0, stream>>>(xb, wqkv, in_proj_b,
                                                              Qb, Kb, Vtb);

  // 4. flash attention with adjacency bias (WMMA + async-to-LDS)
  attn_flash<<<(BG * HH * (NN / 16)) / 4, 128, 0, stream>>>(Qb, Kb, Vtb, adj,
                                                            bias_table, Ob);

  // 5. out_proj + residual -> t1 (WMMA)
  gemm_generic<false><<<dim3(TOT / 64, CC / 64), 128, 0, stream>>>(
      Ob, wob, out_proj_b, x, t1, (__bf16*)nullptr, CC, CC);

  // 6. norm2 -> h2 (f32 + bf16)
  bn_partial<<<64, 256, 0, stream>>>(t1, sums2, sumsq2);
  bn_finalize<<<1, CC, 0, stream>>>(sums2, sumsq2, mu2, rs2);
  bn_apply<<<(TOT * CC + 255) / 256, 256, 0, stream>>>(t1, mu2, rs2, n2g, n2b, h2, h2b);

  // 7. MLP (WMMA): relu(h2 @ w1^T + b1) @ w2^T + b2 + h2 -> t2
  gemm_generic<true><<<dim3(TOT / 64, (2 * CC) / 64), 128, 0, stream>>>(
      h2b, w1b, mlp_b1, (const float*)nullptr, (float*)nullptr, m1b, CC, 2 * CC);
  gemm_generic<false><<<dim3(TOT / 64, CC / 64), 128, 0, stream>>>(
      m1b, w2b, mlp_b2, h2, t2, (__bf16*)nullptr, 2 * CC, CC);

  // 8. norm3 -> final output
  bn_partial<<<64, 256, 0, stream>>>(t2, sums3, sumsq3);
  bn_finalize<<<1, CC, 0, stream>>>(sums3, sumsq3, mu3, rs3);
  bn_apply<<<(TOT * CC + 255) / 256, 256, 0, stream>>>(t2, mu3, rs3, n3g, n3b, out,
                                                       (__bf16*)nullptr);
}
